// C_VAE_NET_9320079032865
// MI455X (gfx1250) — matrix operand order repaired and checked by reading, not compiled, not run on hardware
//
#include <hip/hip_runtime.h>
#include <hip/hip_bf16.h>
#include <math.h>

// ---------------------------------------------------------------------------
// C-VAE RNN (B=256, T=512, D=128, Z=64, E=256, Dc=256) as a persistent
// wave32 WMMA kernel for gfx1250.
//
//  * 16 workgroups, each owns a 16-row batch tile and iterates all T steps.
//  * Recurrent state lives in LDS as bf16 row-major, double-buffered for
//    h_enc/h_dec so only 5 barriers per step are needed.
//  * A fragments: 2x ds_load_b128, software-pipelined one K-block ahead.
//  * Weights pre-packed (f32 -> bf16, WMMA B-fragment order) into d_ws;
//    the compiler keeps them register/L2 resident across the T loop.
//  * Matrix math: v_wmma_f32_16x16x32_bf16, f32 accumulation.
//  * Nonlinearities: hardware v_tanh_f32 when available, v_exp/v_rcp else.
// ---------------------------------------------------------------------------

typedef __attribute__((ext_vector_type(16))) __bf16 v16bf;
typedef __attribute__((ext_vector_type(8)))  float  v8f;
typedef __attribute__((ext_vector_type(4)))  unsigned int v4u;

#define B_  256
#define T_  512
#define D_  128
#define Z_  64
#define E_  256
#define DC_ 256

__device__ __forceinline__ float fast_rcp(float x) {
#if __has_builtin(__builtin_amdgcn_rcpf)
    return __builtin_amdgcn_rcpf(x);
#else
    return 1.f / x;
#endif
}

__device__ __forceinline__ float fast_tanh(float x) {
#if __has_builtin(__builtin_amdgcn_tanhf)
    return __builtin_amdgcn_tanhf(x);        // gfx1250 v_tanh_f32 (TRANS)
#elif __has_builtin(__builtin_amdgcn_tanh_f32)
    return __builtin_amdgcn_tanh_f32(x);
#else
    const float t = __expf(2.f * x);
    return (t - 1.f) * fast_rcp(t + 1.f);
#endif
}

__device__ __forceinline__ float fast_sigmoid(float x) {
    return fast_rcp(1.f + __expf(-x));
}

__device__ __forceinline__ unsigned short f32_bf16(float f) {
    union { float f; unsigned u; } v; v.f = f;
    const unsigned u = v.u;
    return (unsigned short)((u + 0x7FFFu + ((u >> 16) & 1u)) >> 16);  // RNE
}

__device__ __forceinline__ v16bf load_frag2(const unsigned short* p0,
                                            const unsigned short* p1) {
    union { v16bf v; v4u u[2]; } r;
    r.u[0] = *(const v4u*)p0;
    r.u[1] = *(const v4u*)p1;
    return r.v;
}

// A fragment (16x32 bf16) from row-major LDS [16][stride].
// ISA 16-bit A layout: lane l, elem j -> row = l&15,
//   col = k0 + 8*(l>>4) + j            (j<8)
//   col = k0 + 16 + 8*(l>>4) + (j-8)   (j>=8)
__device__ __forceinline__ v16bf load_afrag(const unsigned short* base, int stride,
                                            int k0, int lane) {
    const unsigned short* p = base + (lane & 15) * stride + k0 + 8 * (lane >> 4);
    return load_frag2(p, p + 16);
}

// B fragment from pre-packed weights (32 lanes x 16 bf16 contiguous per frag)
__device__ __forceinline__ v16bf load_bfrag(const unsigned short* wp, int NT,
                                            int kb, int nt, int lane) {
    const unsigned short* p = wp + ((((kb * NT) + nt) * 32 + lane) << 4);
    return load_frag2(p, p + 8);
}

__device__ __forceinline__ v8f wmma_bf16(v16bf a, v16bf b, v8f c) {
    return __builtin_amdgcn_wmma_f32_16x16x32_bf16(false, a, false, b,
                                                   (short)0, c, false, false);
}

// ---------------------------------------------------------------------------
// Weight pre-pack: W[K][N] f32 row-major -> bf16 fragments in WMMA B order.
// out flat index = ((kb*NT + nt)*32 + lane)*16 + j
//   src: k = kb*32 + 8*(lane>>4) + (j&7) + 16*(j>>3), n = nt*16 + (lane&15)
// ---------------------------------------------------------------------------
__global__ void pack_w(const float* __restrict__ W, int K, int N,
                       unsigned short* __restrict__ out) {
    int idx = blockIdx.x * blockDim.x + threadIdx.x;
    if (idx >= K * N) return;
    int j    = idx & 15;
    int lane = (idx >> 4) & 31;
    int fi   = idx >> 9;
    int NT   = N >> 4;
    int kb   = fi / NT;
    int nt   = fi - kb * NT;
    int k = kb * 32 + 8 * (lane >> 4) + (j & 7) + 16 * (j >> 3);
    int n = nt * 16 + (lane & 15);
    out[idx] = f32_bf16(W[k * N + n]);
}

// ---------------------------------------------------------------------------
// Persistent RNN kernel: grid=16 (one 16-row batch tile each), block=256 (8 waves)
// ---------------------------------------------------------------------------
__global__ __launch_bounds__(256)
void vae_rnn_kernel(const float* __restrict__ x, const float* __restrict__ e,
                    const float* __restrict__ b_enc, const float* __restrict__ b_mu,
                    const float* __restrict__ b_sigma, const float* __restrict__ b_dec,
                    const float* __restrict__ b_hdec,
                    const unsigned short* __restrict__ wp_enc_ih,
                    const unsigned short* __restrict__ wp_enc_hh,
                    const unsigned short* __restrict__ wp_mu,
                    const unsigned short* __restrict__ wp_sigma,
                    const unsigned short* __restrict__ wp_dec_ih,
                    const unsigned short* __restrict__ wp_dec_hh,
                    const unsigned short* __restrict__ wp_hdec,
                    float* __restrict__ out) {
    __shared__ unsigned short s_xinp[16 * 256];     // [x_t | x_t - csig] bf16
    __shared__ unsigned short s_henc[2][16 * 256];  // h_enc bf16, ping-pong
    __shared__ unsigned short s_hdec[2][16 * 256];  // h_dec bf16, ping-pong
    __shared__ unsigned short s_z[16 * 64];         // z bf16
    __shared__ float          s_sig[16 * 64];       // sigma f32 handoff

    const int tid  = threadIdx.x;
    const int w    = tid >> 5;
    const int lane = tid & 31;
    const int half = lane >> 4;
    const int ln   = lane & 15;
    const int m0   = blockIdx.x * 16;

    float* out_xhat = out;                                   // [256][128]
    float* out_mu   = out + B_ * D_;                         // [256][512][64]
    float* out_ls   = out_mu  + (size_t)B_ * T_ * Z_;
    float* out_sig  = out_ls  + (size_t)B_ * T_ * Z_;
    float* out_z    = out_sig + (size_t)B_ * T_ * Z_;

    // zero recurrent state (both buffers)
    for (int i = tid; i < 2 * 16 * 256; i += 256) {
        s_henc[0][i] = 0; s_hdec[0][i] = 0;   // flat across both buffers
    }
    for (int i = tid; i < 16 * 64; i += 256) s_z[i] = 0;

    // constant per-wave bias / e preloads
    const float be0 = b_enc[(2 * w + 0) * 16 + ln];
    const float be1 = b_enc[(2 * w + 1) * 16 + ln];
    const int   zcol = (w & 3) * 16 + ln;
    const float bms = (w < 4) ? b_mu[zcol] : b_sigma[zcol];
    const float bd0 = b_dec[(2 * w + 0) * 16 + ln];
    const float bd1 = b_dec[(2 * w + 1) * 16 + ln];
    const float bh  = b_hdec[w * 16 + ln];

    float e_r[8];
    #pragma unroll
    for (int r = 0; r < 8; ++r)
        e_r[r] = e[(size_t)(m0 + r + 8 * half) * Z_ + zcol];

    float c_state = 0.f;                 // iterated-sigmoid carry (scalar)
    const int xm = tid >> 4;             // x-staging: row 0..15
    const int xd = (tid & 15) * 8;       // x-staging: col base

    __syncthreads();

    for (int t = 0; t < T_; ++t) {
        const int pp = t & 1;            // read buffer
        const int qq = pp ^ 1;           // write buffer
        const float csig = fast_sigmoid(c_state);

        // ---- stage x_t as bf16 [x | x - csig] ------------------------------
        {
            const float* xr = x + ((size_t)(m0 + xm) * T_ + t) * D_ + xd;
            float4 a = *(const float4*)xr;
            float4 b = *(const float4*)(xr + 4);
            float v[8] = {a.x, a.y, a.z, a.w, b.x, b.y, b.z, b.w};
            #pragma unroll
            for (int i = 0; i < 8; ++i) {
                s_xinp[xm * 256 + xd + i]       = f32_bf16(v[i]);
                s_xinp[xm * 256 + 128 + xd + i] = f32_bf16(v[i] - csig);
            }
            if (t + 1 < T_) __builtin_prefetch(xr + D_, 0, 3);  // next x_t
        }
        __syncthreads();                                        // (1)

        // ---- encoder: h_enc' = tanh([x|x^]@W_ih + h_enc@W_hh + b), K=512 ---
        v8f acc0, acc1;
        #pragma unroll
        for (int r = 0; r < 8; ++r) { acc0[r] = be0; acc1[r] = be1; }
        {
            v16bf a_cur = load_afrag(s_xinp, 256, 0, lane);
            #pragma unroll
            for (int kb = 0; kb < 16; ++kb) {
                v16bf a_nxt = a_cur;                  // pipeline: load kb+1 early
                if (kb + 1 < 16)
                    a_nxt = (kb + 1 < 8)
                          ? load_afrag(s_xinp, 256, (kb + 1) * 32, lane)
                          : load_afrag(s_henc[pp], 256, (kb + 1 - 8) * 32, lane);
                const unsigned short* wb = (kb < 8) ? wp_enc_ih : wp_enc_hh;
                v16bf b0 = load_bfrag(wb, 16, kb & 7, 2 * w + 0, lane);
                v16bf b1 = load_bfrag(wb, 16, kb & 7, 2 * w + 1, lane);
                acc0 = wmma_bf16(a_cur, b0, acc0);
                acc1 = wmma_bf16(a_cur, b1, acc1);
                a_cur = a_nxt;
            }
        }
        // write new h_enc into the other buffer (no pre-barrier needed)
        #pragma unroll
        for (int r = 0; r < 8; ++r) {
            const int M = r + 8 * half;
            s_henc[qq][M * 256 + (2 * w + 0) * 16 + ln] = f32_bf16(fast_tanh(acc0[r]));
            s_henc[qq][M * 256 + (2 * w + 1) * 16 + ln] = f32_bf16(fast_tanh(acc1[r]));
        }
        __syncthreads();                                        // (2)

        // ---- mu (waves 0-3) / logsigma (waves 4-7): h_enc' @ w, K=256 ------
        v8f accz;
        #pragma unroll
        for (int r = 0; r < 8; ++r) accz[r] = bms;
        {
            const unsigned short* wms = (w < 4) ? wp_mu : wp_sigma;
            v16bf a_cur = load_afrag(s_henc[qq], 256, 0, lane);
            #pragma unroll
            for (int kb = 0; kb < 8; ++kb) {
                v16bf a_nxt = a_cur;
                if (kb + 1 < 8)
                    a_nxt = load_afrag(s_henc[qq], 256, (kb + 1) * 32, lane);
                v16bf bb = load_bfrag(wms, 4, kb, (w & 3), lane);
                accz = wmma_bf16(a_cur, bb, accz);
                a_cur = a_nxt;
            }
        }
        if (w >= 4) {                    // sigma waves
            #pragma unroll
            for (int r = 0; r < 8; ++r) {
                const int M = r + 8 * half;
                const size_t o = ((size_t)(m0 + M) * T_ + t) * Z_ + zcol;
                const float sg = __expf(accz[r]);
                out_ls[o]  = accz[r];
                out_sig[o] = sg;
                s_sig[M * 64 + zcol] = sg;
            }
        } else {                         // mu waves
            #pragma unroll
            for (int r = 0; r < 8; ++r) {
                const int M = r + 8 * half;
                out_mu[((size_t)(m0 + M) * T_ + t) * Z_ + zcol] = accz[r];
            }
        }
        __syncthreads();                                        // (3)
        if (w < 4) {                     // z = mu + sigma * e
            #pragma unroll
            for (int r = 0; r < 8; ++r) {
                const int M = r + 8 * half;
                const float zv = accz[r] + s_sig[M * 64 + zcol] * e_r[r];
                out_z[((size_t)(m0 + M) * T_ + t) * Z_ + zcol] = zv;
                s_z[M * 64 + zcol] = f32_bf16(zv);
            }
        }
        __syncthreads();                                        // (4)

        // ---- decoder: h_dec' = tanh(z@W_ih + h_dec@W_hh + b), K=320 --------
        v8f d0, d1;
        #pragma unroll
        for (int r = 0; r < 8; ++r) { d0[r] = bd0; d1[r] = bd1; }
        {
            v16bf a_cur = load_afrag(s_z, 64, 0, lane);
            #pragma unroll
            for (int kb = 0; kb < 10; ++kb) {
                v16bf a_nxt = a_cur;
                if (kb + 1 < 10)
                    a_nxt = (kb + 1 < 2)
                          ? load_afrag(s_z, 64, (kb + 1) * 32, lane)
                          : load_afrag(s_hdec[pp], 256, (kb + 1 - 2) * 32, lane);
                const unsigned short* wb = (kb < 2) ? wp_dec_ih : wp_dec_hh;
                const int kbb = (kb < 2) ? kb : kb - 2;
                v16bf b0 = load_bfrag(wb, 16, kbb, 2 * w + 0, lane);
                v16bf b1 = load_bfrag(wb, 16, kbb, 2 * w + 1, lane);
                d0 = wmma_bf16(a_cur, b0, d0);
                d1 = wmma_bf16(a_cur, b1, d1);
                a_cur = a_nxt;
            }
        }
        #pragma unroll
        for (int r = 0; r < 8; ++r) {
            const int M = r + 8 * half;
            s_hdec[qq][M * 256 + (2 * w + 0) * 16 + ln] = f32_bf16(fast_tanh(d0[r]));
            s_hdec[qq][M * 256 + (2 * w + 1) * 16 + ln] = f32_bf16(fast_tanh(d1[r]));
        }
        __syncthreads();                                        // (5)

        c_state = csig;
    }

    // ---- last step only: x_hat_dec = h_dec @ w_h_dec + b + x_{T-1} ---------
    const int fb = ((T_ - 1) & 1) ^ 1;   // buffer holding final h_dec
    v8f f;
    #pragma unroll
    for (int r = 0; r < 8; ++r) f[r] = bh;
    {
        v16bf a_cur = load_afrag(s_hdec[fb], 256, 0, lane);
        #pragma unroll
        for (int kb = 0; kb < 8; ++kb) {
            v16bf a_nxt = a_cur;
            if (kb + 1 < 8)
                a_nxt = load_afrag(s_hdec[fb], 256, (kb + 1) * 32, lane);
            v16bf bb = load_bfrag(wp_hdec, 8, kb, w, lane);
            f = wmma_bf16(a_cur, bb, f);
            a_cur = a_nxt;
        }
    }
    const int col = w * 16 + ln;
    #pragma unroll
    for (int r = 0; r < 8; ++r) {
        const int M = r + 8 * half;
        const float xv = x[((size_t)(m0 + M) * T_ + (T_ - 1)) * D_ + col];
        out_xhat[(m0 + M) * D_ + col] = f[r] + xv;
    }
}

// ---------------------------------------------------------------------------
extern "C" void kernel_launch(void* const* d_in, const int* in_sizes, int n_in,
                              void* d_out, int out_size, void* d_ws, size_t ws_size,
                              hipStream_t stream) {
    const float* x        = (const float*)d_in[0];
    const float* e        = (const float*)d_in[1];
    const float* W_enc_ih = (const float*)d_in[2];
    const float* W_enc_hh = (const float*)d_in[3];
    const float* b_enc    = (const float*)d_in[4];
    const float* W_dec_ih = (const float*)d_in[5];
    const float* W_dec_hh = (const float*)d_in[6];
    const float* b_dec    = (const float*)d_in[7];
    const float* w_mu     = (const float*)d_in[8];
    const float* b_mu     = (const float*)d_in[9];
    const float* w_sigma  = (const float*)d_in[10];
    const float* b_sigma  = (const float*)d_in[11];
    const float* w_h_dec  = (const float*)d_in[12];
    const float* b_h_dec  = (const float*)d_in[13];

    unsigned short* ws        = (unsigned short*)d_ws;
    unsigned short* wp_enc_ih = ws;                      // 256x256
    unsigned short* wp_enc_hh = wp_enc_ih + 256 * 256;   // 256x256
    unsigned short* wp_mu     = wp_enc_hh + 256 * 256;   // 256x64
    unsigned short* wp_sigma  = wp_mu     + 256 * 64;    // 256x64
    unsigned short* wp_dec_ih = wp_sigma  + 256 * 64;    // 64x256
    unsigned short* wp_dec_hh = wp_dec_ih + 64 * 256;    // 256x256
    unsigned short* wp_hdec   = wp_dec_hh + 256 * 256;   // 256x128

    auto packLaunch = [&](const float* W, int K, int N, unsigned short* o) {
        const int total = K * N;
        pack_w<<<(total + 255) / 256, 256, 0, stream>>>(W, K, N, o);
    };
    packLaunch(W_enc_ih, 256, 256, wp_enc_ih);
    packLaunch(W_enc_hh, 256, 256, wp_enc_hh);
    packLaunch(w_mu,     256,  64, wp_mu);
    packLaunch(w_sigma,  256,  64, wp_sigma);
    packLaunch(W_dec_ih,  64, 256, wp_dec_ih);
    packLaunch(W_dec_hh, 256, 256, wp_dec_hh);
    packLaunch(w_h_dec,  256, 128, wp_hdec);

    vae_rnn_kernel<<<16, 256, 0, stream>>>(
        x, e, b_enc, b_mu, b_sigma, b_dec, b_h_dec,
        wp_enc_ih, wp_enc_hh, wp_mu, wp_sigma, wp_dec_ih, wp_dec_hh, wp_hdec,
        (float*)d_out);
}
